// BidirectionalLayerFuse_47373489274952
// MI455X (gfx1250) — compile-verified, matrix operand order checked by reading, not run on hardware
//
#include <hip/hip_runtime.h>
#include <hip/hip_bf16.h>

typedef __attribute__((ext_vector_type(16))) _Float16 v16h;
typedef __attribute__((ext_vector_type(8)))  float    v8f;

#define NPTS 8192
#define NBAT 2
#define CCH  64
#define KNNK 16
#define LEAKYF 0.1f
#define CHUNK 256            // candidates staged per LDS chunk in KNN

__device__ __forceinline__ float leakyf(float x) { return x >= 0.f ? x : LEAKYF * x; }

// ISA 7.12.2 16-bit A/B fragment K index: lane holds K = kb + {0..7, 16..23},
// kb = (lane>>4)*8 ; ext-vector element i -> K = kb + (i<8 ? i : i+8)
__device__ __forceinline__ int kmap(int lane, int i) {
  return ((lane >> 4) << 3) + (i < 8 ? i : i + 8);
}

// --- async global->LDS staging (gfx1250 GLOBAL_LOAD_ASYNC_TO_LDS_B32) -------
#if __has_builtin(__builtin_amdgcn_global_load_async_to_lds_b32)
#define ASYNC_STAGE 1
#else
#define ASYNC_STAGE 0
#endif

__device__ __forceinline__ void stage_f32(float* lds_dst, const float* gsrc) {
#if ASYNC_STAGE
  __builtin_amdgcn_global_load_async_to_lds_b32(
      (__attribute__((address_space(1))) int*)gsrc,
      (__attribute__((address_space(3))) int*)lds_dst, 0, 0);
#else
  *lds_dst = *gsrc;
#endif
}

__device__ __forceinline__ void stage_wait() {
#if ASYNC_STAGE
#if __has_builtin(__builtin_amdgcn_s_wait_asynccnt)
  __builtin_amdgcn_s_wait_asynccnt(0);
#else
  asm volatile("s_wait_asynccnt 0" ::: "memory");
#endif
#endif
}

// ---------------------------------------------------------------------------
// Kernel 1: four 1x1-conv projections, WMMA f16 -> f32.
//   combo 0: P0 = W11*feat1+b11   combo 1: P1 = W22*feat2+b22
//   combo 2: P2 = W11*feat2+b11   combo 3: P3 = W22*feat1+b22
// Output layout P[combo][b][n][c]  (channel-last for gathers).
// ---------------------------------------------------------------------------
__global__ void __launch_bounds__(32)
proj_kernel(const float* __restrict__ feat1, const float* __restrict__ feat2,
            const float* __restrict__ W11, const float* __restrict__ b11,
            const float* __restrict__ W22, const float* __restrict__ b22,
            float* __restrict__ P)
{
  const int tile  = blockIdx.x;              // 0 .. NBAT*(NPTS/16)-1
  const int combo = blockIdx.y;              // 0..3
  const int b  = tile / (NPTS / 16);
  const int n0 = (tile % (NPTS / 16)) * 16;
  const float* feat = (combo == 0 || combo == 3) ? feat1 : feat2;
  const float* W    = (combo == 0 || combo == 2) ? W11   : W22;
  const float* bias = (combo == 0 || combo == 2) ? b11   : b22;
  feat += (size_t)b * CCH * NPTS;

  const int lane = threadIdx.x;
  const int m    = lane & 15;                // matrix row (point) this lane holds

  // A fragments for ksteps 0,1:  A[m][k] = feat[k][n0+m]
  v16h a0, a1;
#pragma unroll
  for (int i = 0; i < 16; ++i) {
    const int k = kmap(lane, i);
    a0[i] = (_Float16)feat[(size_t)(k)      * NPTS + n0 + m];
    a1[i] = (_Float16)feat[(size_t)(k + 32) * NPTS + n0 + m];
  }

  const size_t base = (((size_t)combo * NBAT + b) * NPTS) * CCH;
#pragma unroll
  for (int ct = 0; ct < 4; ++ct) {
    v16h bf0, bf1;                           // B[k][n] = W[n][k], n = out channel
#pragma unroll
    for (int i = 0; i < 16; ++i) {
      const int k = kmap(lane, i);
      const int o = ct * 16 + m;
      bf0[i] = (_Float16)W[(size_t)o * CCH + k];
      bf1[i] = (_Float16)W[(size_t)o * CCH + k + 32];
    }
    v8f acc = {};
    acc = __builtin_amdgcn_wmma_f32_16x16x32_f16(false, a0, false, bf0,
                                                 (short)0, acc, false, false);
    acc = __builtin_amdgcn_wmma_f32_16x16x32_f16(false, a1, false, bf1,
                                                 (short)0, acc, false, false);
    const int o  = ct * 16 + m;              // D: lane holds column n = lane&15
    const float bv = bias[o];
#pragma unroll
    for (int r = 0; r < 8; ++r) {
      const int mm = r + 8 * (lane >> 4);    // D row
      P[base + (size_t)(n0 + mm) * CCH + o] = acc[r] + bv;
    }
  }
}

// ---------------------------------------------------------------------------
// Kernel 2: 6-D KNN, top-16 of 8192 per query. One wave per query,
// 8 consecutive queries (same dir,b) per block sharing the candidate set.
// Candidates double-buffer staged into LDS via async global->LDS loads
// (ASYNCcnt) so each block reads the candidate arrays once, not 8x.
// idx_out[dir][b][q][k]
// ---------------------------------------------------------------------------
__global__ void __launch_bounds__(256)
knn_kernel(const float* __restrict__ pc1, const float* __restrict__ pc2,
           const float* __restrict__ kn1, const float* __restrict__ kn2,
           int* __restrict__ idx_out)
{
  const int tid  = threadIdx.x;
  const int w    = tid >> 5;
  const int lane = tid & 31;
  const int gq   = blockIdx.x * 8;           // first query slot of this block
  const int dir = gq / (NBAT * NPTS);
  const int rem = gq % (NBAT * NPTS);
  const int b   = rem / NPTS;
  const int q   = (rem % NPTS) + w;          // this wave's query

  const float* qp = dir == 0 ? pc1 : pc2;
  const float* qk = dir == 0 ? kn1 : kn2;
  const float* cp = dir == 0 ? pc2 : pc1;
  const float* ck = dir == 0 ? kn2 : kn1;
  const size_t pb = (size_t)b * 3 * NPTS;

  const float q0 = qp[pb + 0 * NPTS + q], q1 = qp[pb + 1 * NPTS + q];
  const float q2 = qp[pb + 2 * NPTS + q];
  const float q3 = qk[pb + 0 * NPTS + q], q4 = qk[pb + 1 * NPTS + q];
  const float q5 = qk[pb + 2 * NPTS + q];

  __shared__ float cand[2][6 * CHUNK];       // [buf][comp*CHUNK + i]

  float dl[KNNK]; int il[KNNK];
#pragma unroll
  for (int j = 0; j < KNNK; ++j) { dl[j] = 3.4e38f; il[j] = 0; }

  // issue one chunk: thread tid stages candidate c*CHUNK+tid (6 components)
  auto issue = [&](int c, int buf) {
    const int cc = c * CHUNK + tid;
#pragma unroll
    for (int j = 0; j < 3; ++j) {
      stage_f32(&cand[buf][j * CHUNK + tid],       &cp[pb + (size_t)j * NPTS + cc]);
      stage_f32(&cand[buf][(3 + j) * CHUNK + tid], &ck[pb + (size_t)j * NPTS + cc]);
    }
  };

  issue(0, 0);
  for (int c = 0; c < NPTS / CHUNK; ++c) {
    const int cur = c & 1;
    stage_wait();                            // my wave's staged loads landed
    __syncthreads();                         // every wave's loads landed
    if (c + 1 < NPTS / CHUNK) issue(c + 1, cur ^ 1);

    for (int i = lane; i < CHUNK; i += 32) {
      const float d0 = cand[cur][0 * CHUNK + i] - q0;
      const float d1 = cand[cur][1 * CHUNK + i] - q1;
      const float d2 = cand[cur][2 * CHUNK + i] - q2;
      const float d3 = cand[cur][3 * CHUNK + i] - q3;
      const float d4 = cand[cur][4 * CHUNK + i] - q4;
      const float d5 = cand[cur][5 * CHUNK + i] - q5;
      const float d  = d0 * d0 + d1 * d1 + d2 * d2 +
                       d3 * d3 + d4 * d4 + d5 * d5;
      if (d < dl[KNNK - 1]) {
        float dv = d; int iv = c * CHUNK + i;
#pragma unroll
        for (int j = 0; j < KNNK; ++j) {
          if (dv < dl[j]) {
            float tf = dl[j]; dl[j] = dv; dv = tf;
            int   ti = il[j]; il[j] = iv; iv = ti;
          }
        }
      }
    }
    __syncthreads();                         // buffer free before re-stage
  }

  // dump per-lane sorted lists to LDS, then 16-round k-way merge
  __shared__ float sd[8][32 * KNNK];
  __shared__ int   si[8][32 * KNNK];
#pragma unroll
  for (int j = 0; j < KNNK; ++j) {
    sd[w][lane * KNNK + j] = dl[j];
    si[w][lane * KNNK + j] = il[j];
  }
  __syncthreads();

  int pos = 0, myidx = 0;
  for (int r = 0; r < KNNK; ++r) {
    float candd = pos < KNNK ? sd[w][lane * KNNK + pos] : 3.4e38f;
    int   ci    = pos < KNNK ? si[w][lane * KNNK + pos] : 0;
    int   cl    = lane;
#pragma unroll
    for (int off = 16; off > 0; off >>= 1) {
      float od = __shfl_xor(candd, off, 32);
      int   oi = __shfl_xor(ci,    off, 32);
      int   ol = __shfl_xor(cl,    off, 32);
      if (od < candd || (od == candd && ol < cl)) { candd = od; ci = oi; cl = ol; }
    }
    if (lane == cl) pos++;
    if (lane == r)  myidx = ci;
  }
  if (lane < KNNK)
    idx_out[(((size_t)dir * NBAT + b) * NPTS + q) * KNNK + lane] = myidx;
}

// ---------------------------------------------------------------------------
// Kernel 3: fused gather + pos-encode + add + leaky + (64x64 MLP via WMMA)
//           + leaky + max-over-K, direct to transposed output.
// Block = 8 waves = one 16-query tile. Wave w owns queries 2w, 2w+1;
// each row-tile (16 rows) = the 16 neighbors of one query, so the K-max
// is a max over M of the WMMA accumulator.
// ---------------------------------------------------------------------------
__global__ void __launch_bounds__(256)
cross_kernel(const float* __restrict__ pc1, const float* __restrict__ pc2,
             const float* __restrict__ P, const int* __restrict__ idx,
             const float* __restrict__ Wpos, const float* __restrict__ bpos,
             const float* __restrict__ Wm,   const float* __restrict__ bm,
             float* __restrict__ out)
{
  const int blk = blockIdx.x;
  const int dir = blk / (NBAT * (NPTS / 16));
  const int rem = blk % (NBAT * (NPTS / 16));
  const int b   = rem / (NPTS / 16);
  const int q0  = (rem % (NPTS / 16)) * 16;
  const int w    = threadIdx.x >> 5;
  const int lane = threadIdx.x & 31;

  const float* xyzq = dir == 0 ? pc1 : pc2;
  const float* xyzc = dir == 0 ? pc2 : pc1;
  const float* P1 = P + (((size_t)(dir == 0 ? 0 : 2)) * NBAT + b) * NPTS * CCH;
  const float* P2 = P + (((size_t)(dir == 0 ? 1 : 3)) * NBAT + b) * NPTS * CCH;
  const int* idxb = idx + (((size_t)dir * NBAT + b) * NPTS) * KNNK;

  __shared__ _Float16 Anew[256][CCH + 2];    // +2 halves pad vs bank conflicts

  // ---- build 'new' rows (one lane per row; wave owns 32 rows = 2 queries)
  {
    const int qlb = 2 * w + (lane >> 4);     // query-in-block 0..15
    const int kk  = lane & 15;               // neighbor slot
    const int qg  = q0 + qlb;
    const int nb  = idxb[(size_t)qg * KNNK + kk];
    const size_t pb = (size_t)b * 3 * NPTS;
    const float dx = xyzc[pb + 0 * NPTS + nb] - xyzq[pb + 0 * NPTS + qg];
    const float dy = xyzc[pb + 1 * NPTS + nb] - xyzq[pb + 1 * NPTS + qg];
    const float dz = xyzc[pb + 2 * NPTS + nb] - xyzq[pb + 2 * NPTS + qg];
    const float4* g2 = (const float4*)(P2 + (size_t)nb * CCH);
    const float4* p1 = (const float4*)(P1 + (size_t)qg * CCH);
    const int row = qlb * 16 + kk;
#pragma unroll 4
    for (int c4 = 0; c4 < CCH / 4; ++c4) {
      const float4 g = g2[c4];
      const float4 p = p1[c4];
      const float gv[4] = {g.x, g.y, g.z, g.w};
      const float pv[4] = {p.x, p.y, p.z, p.w};
#pragma unroll
      for (int j = 0; j < 4; ++j) {
        const int c = c4 * 4 + j;
        float pos = Wpos[c * 3 + 0] * dx + Wpos[c * 3 + 1] * dy +
                    Wpos[c * 3 + 2] * dz + bpos[c];
        Anew[row][c] = (_Float16)leakyf(gv[j] + pv[j] + pos);
      }
    }
  }

  // ---- B fragments of Wm (B[k][n] = Wm[n][k]), 2 ksteps x 4 col-tiles
  v16h bf[2][4];
#pragma unroll
  for (int ct = 0; ct < 4; ++ct)
#pragma unroll
    for (int i = 0; i < 16; ++i) {
      const int n = ct * 16 + (lane & 15);
      const int k = kmap(lane, i);
      bf[0][ct][i] = (_Float16)Wm[(size_t)n * CCH + k];
      bf[1][ct][i] = (_Float16)Wm[(size_t)n * CCH + k + 32];
    }

  __syncthreads();

  // ---- WMMA + bias + leaky + max-over-K (rows) per query
  const size_t obase = ((size_t)dir * NBAT + b) * CCH * NPTS;
#pragma unroll
  for (int rt = 0; rt < 2; ++rt) {
    const int qlb = 2 * w + rt;
    v16h a0, a1;                              // A[m][k] from LDS rows of query qlb
#pragma unroll
    for (int i = 0; i < 16; ++i) {
      const int k = kmap(lane, i);
      a0[i] = Anew[qlb * 16 + (lane & 15)][k];
      a1[i] = Anew[qlb * 16 + (lane & 15)][k + 32];
    }
#pragma unroll
    for (int ct = 0; ct < 4; ++ct) {
      v8f acc = {};
      acc = __builtin_amdgcn_wmma_f32_16x16x32_f16(false, a0, false, bf[0][ct],
                                                   (short)0, acc, false, false);
      acc = __builtin_amdgcn_wmma_f32_16x16x32_f16(false, a1, false, bf[1][ct],
                                                   (short)0, acc, false, false);
      const int o  = ct * 16 + (lane & 15);   // D column = out channel
      const float bv = bm[o];
      float mx = -3.4e38f;
#pragma unroll
      for (int r = 0; r < 8; ++r) mx = fmaxf(mx, leakyf(acc[r] + bv));
      mx = fmaxf(mx, __shfl_xor(mx, 16, 32)); // merge the two M halves
      if (lane < 16)
        out[obase + (size_t)o * NPTS + (q0 + qlb)] = mx;
    }
  }
}

// ---------------------------------------------------------------------------
extern "C" void kernel_launch(void* const* d_in, const int* in_sizes, int n_in,
                              void* d_out, int out_size, void* d_ws, size_t ws_size,
                              hipStream_t stream) {
  const float* pc1   = (const float*)d_in[0];
  const float* pc2   = (const float*)d_in[1];
  const float* feat1 = (const float*)d_in[2];
  const float* feat2 = (const float*)d_in[3];
  const float* kn1   = (const float*)d_in[4];
  const float* kn2   = (const float*)d_in[5];
  const float* Wpos  = (const float*)d_in[6];
  const float* bpos  = (const float*)d_in[7];
  const float* W11   = (const float*)d_in[8];
  const float* b11   = (const float*)d_in[9];
  const float* W22   = (const float*)d_in[10];
  const float* b22   = (const float*)d_in[11];
  const float* Wm    = (const float*)d_in[12];
  const float* bm    = (const float*)d_in[13];

  float* P    = (float*)d_ws;                                  // 4*B*N*64 f32
  int*   idxb = (int*)((char*)d_ws +
                       (size_t)4 * NBAT * NPTS * CCH * sizeof(float));

  dim3 pgrid(NBAT * (NPTS / 16), 4);
  proj_kernel<<<pgrid, 32, 0, stream>>>(feat1, feat2, W11, b11, W22, b22, P);

  knn_kernel<<<(2 * NBAT * NPTS) / 8, 256, 0, stream>>>(pc1, pc2, kn1, kn2, idxb);

  cross_kernel<<<2 * NBAT * (NPTS / 16), 256, 0, stream>>>(
      pc1, pc2, P, idxb, Wpos, bpos, Wm, bm, (float*)d_out);
}
